// SubsetsSampleWeightedFormulaGRUNewSparse_71347996721717
// MI455X (gfx1250) — matrix operand order repaired and checked by reading, not compiled, not run on hardware
//
#include <hip/hip_runtime.h>
#include <hip/hip_bf16.h>
#include <math.h>

// ---------------- constants ----------------
#define BB     16
#define AA     32
#define GG     512
#define SS     1024
#define MM     8
#define EE     5
#define OHH    20
#define FEPAD  128      // FE=100 padded to multiple of 32 for WMMA K loop
#define DD     512
#define NBINS  512
#define G3     1536
#define MROWS  (BB * SS)   // 16384

typedef __attribute__((ext_vector_type(16))) _Float16 v16h;
typedef __attribute__((ext_vector_type(8)))  _Float16 h8;
typedef __attribute__((ext_vector_type(8)))  float    v8f;

// ---------------- block reductions (blockDim.x == 256) ----------------
__device__ __forceinline__ float block_sum(float v, float* red) {
    const int t = threadIdx.x;
    red[t] = v;
    __syncthreads();
    #pragma unroll
    for (int o = 128; o > 0; o >>= 1) {
        if (t < o) red[t] += red[t + o];
        __syncthreads();
    }
    float r = red[0];
    __syncthreads();
    return r;
}

__device__ __forceinline__ float block_max(float v, float* red) {
    const int t = threadIdx.x;
    red[t] = v;
    __syncthreads();
    #pragma unroll
    for (int o = 128; o > 0; o >>= 1) {
        if (t < o) red[t] = fmaxf(red[t], red[t + o]);
        __syncthreads();
    }
    float r = red[0];
    __syncthreads();
    return r;
}

// ---------------- weight f32 -> f16 with K padding ----------------
__global__ void cvt_pad_f16(const float* __restrict__ src, _Float16* __restrict__ dst,
                            int N, int K, int Kp) {
    int i = blockIdx.x * 256 + threadIdx.x;
    if (i >= N * Kp) return;
    int n = i / Kp, k = i - n * Kp;
    dst[i] = (k < K) ? (_Float16)src[(size_t)n * K + k] : (_Float16)0.f;
}

__global__ void zero_f32(float* __restrict__ p, int n) {
    int i = blockIdx.x * 256 + threadIdx.x;
    if (i < n) p[i] = 0.f;
}

// ---------------- fused subset pooling + LN + formula encoding ----------------
// one block (256 thr) per m = b*S + s
__global__ __launch_bounds__(256) void pool_ln_formula(
    const float* __restrict__ vert_feat, const float* __restrict__ vmask,
    const float* __restrict__ elem_oh, const int* __restrict__ subs,
    const float* __restrict__ ln_g, const float* __restrict__ ln_b,
    _Float16* __restrict__ h16, _Float16* __restrict__ form16)
{
    const int m = blockIdx.x;
    const int b = m >> 10;              // S = 1024
    const int t = threadIdx.x;
    __shared__ float w[AA], sf[AA], cnt[EE];
    __shared__ float red[256];
    __shared__ float ssz;

    if (t < AA) {
        float sv = (float)subs[(size_t)m * AA + t];
        sf[t] = sv;
        w[t]  = sv * vmask[b * AA + t];
    }
    __syncthreads();
    if (t == 0) {
        float s = 0.f;
        for (int a = 0; a < AA; ++a) s += w[a];
        ssz = s + 1e-4f;
    }
    if (t < EE) {
        float c = 0.f;
        for (int a = 0; a < AA; ++a)
            c += elem_oh[((size_t)b * AA + a) * EE + t] * sf[a];
        cnt[t] = c;
    }
    __syncthreads();

    // subset-weighted mean of masked vertex features (mask applied twice, as in ref)
    float v[2];
    #pragma unroll
    for (int j = 0; j < 2; ++j) {
        const int g = t + j * 256;
        float acc = 0.f;
        for (int a = 0; a < AA; ++a) {
            float wa = w[a];
            if (wa != 0.f)
                acc += vert_feat[((size_t)b * AA + a) * GG + g] * vmask[b * AA + a] * wa;
        }
        v[j] = acc / ssz;
    }
    // layernorm over G=512
    float mean = block_sum(v[0] + v[1], red) * (1.f / 512.f);
    float d0 = v[0] - mean, d1 = v[1] - mean;
    float var = block_sum(d0 * d0 + d1 * d1, red) * (1.f / 512.f);
    float rstd = rsqrtf(var + 1e-5f);
    h16[(size_t)m * GG + t]       = (_Float16)(d0 * rstd * ln_g[t] + ln_b[t]);
    h16[(size_t)m * GG + t + 256] = (_Float16)(d1 * rstd * ln_g[t + 256] + ln_b[t + 256]);

    // thermometer formula encoding (padded to 128)
    if (t < FEPAD) {
        float f = 0.f;
        if (t < EE * OHH) {
            int e = t / OHH, o = t - e * OHH;
            f = ((float)o < cnt[e]) ? 1.f : 0.f;
        }
        form16[(size_t)m * FEPAD + t] = (_Float16)f;
    }
}

// ---------------- WMMA GEMM, register-blocked 16x64 per wave ----------------
// C[gridX*16 x N] = A[. x K] * W[N x K]^T, 4 waves/block -> 16x256 per block.
// One A fragment is reused across 4 accumulators (4 N-tiles) per k-step:
// 2.5 b128 loads per v_wmma instead of 4, and A activation traffic cut 4x.
// Fragment addressing per ISA 7.12.2 (wave32):
//   A 16x32 f16: lane -> M=lane&15; halves 0..7 -> K=k0+(lane>>4)*8..+7, halves 8..15 -> +16
//   B 32x16 f16: lane -> N=lane&15; halves 0..7 -> K=k0+(lane>>4)*8..+7, halves 8..15 -> +16
//   C/D f32:     VGPR i -> M = i + 8*(lane>>4), N = lane&15
__global__ __launch_bounds__(128) void gemm16x4(
    const _Float16* __restrict__ A, const _Float16* __restrict__ W,
    float* __restrict__ Cf, _Float16* __restrict__ Ch,
    const float* __restrict__ bias,
    int N, int K, int relu)
{
    const int lane  = threadIdx.x & 31;
    const int wave  = threadIdx.x >> 5;
    const int mBase = blockIdx.x * 16;
    const int nBase = (blockIdx.y * 4 + wave) * 64;
    const int grp   = lane >> 4;     // 0 or 1
    const int l15   = lane & 15;     // M row for A frag, N col for B frag

    const _Float16* arow = A + (size_t)(mBase + l15) * K;
    const _Float16* wrow = W + (size_t)(nBase + l15) * K;

    v8f acc[4] = {{}, {}, {}, {}};
    for (int k0 = 0; k0 < K; k0 += 32) {
        h8 alo = *(const h8*)(arow + k0 + grp * 8);
        h8 ahi = *(const h8*)(arow + k0 + 16 + grp * 8);
        v16h a;
        #pragma unroll
        for (int i = 0; i < 8; ++i) { a[i] = alo[i]; a[i + 8] = ahi[i]; }
        #pragma unroll
        for (int j = 0; j < 4; ++j) {
            h8 blo = *(const h8*)(wrow + (size_t)j * 16 * K + k0 + grp * 8);
            h8 bhi = *(const h8*)(wrow + (size_t)j * 16 * K + k0 + 16 + grp * 8);
            v16h bf;
            #pragma unroll
            for (int i = 0; i < 8; ++i) { bf[i] = blo[i]; bf[i + 8] = bhi[i]; }
            acc[j] = __builtin_amdgcn_wmma_f32_16x16x32_f16(
                         /*neg_a=*/false, a, /*neg_b=*/false, bf,
                         /*c_mod=*/(short)0, acc[j], /*reuse_a=*/false, /*reuse_b=*/false);
        }
    }

    #pragma unroll
    for (int j = 0; j < 4; ++j) {
        const int n = nBase + j * 16 + l15;
        const float bv = bias ? bias[n] : 0.f;
        #pragma unroll
        for (int i = 0; i < 8; ++i) {
            const int mrow = mBase + grp * 8 + i;
            float val = acc[j][i] + bv;
            if (relu) val = fmaxf(val, 0.f);
            const size_t off = (size_t)mrow * N + n;
            if (Cf) Cf[off] = val;
            if (Ch) Ch[off] = (_Float16)val;
        }
    }
}

// ---------------- fused GRU cell + post layernorm ----------------
__global__ __launch_bounds__(256) void gru_ln(
    const _Float16* __restrict__ gI, const _Float16* __restrict__ gH,
    const _Float16* __restrict__ h16,
    const float* __restrict__ b_ih, const float* __restrict__ b_hh,
    const float* __restrict__ ln_g, const float* __restrict__ ln_b,
    _Float16* __restrict__ x16)
{
    const int m = blockIdx.x, t = threadIdx.x;
    __shared__ float red[256];
    float comb[2];
    #pragma unroll
    for (int j = 0; j < 2; ++j) {
        const int g = t + j * 256;
        const size_t base = (size_t)m * G3;
        float ir = (float)gI[base + g]            + b_ih[g];
        float hr = (float)gH[base + g]            + b_hh[g];
        float iz = (float)gI[base + GG + g]       + b_ih[GG + g];
        float hz = (float)gH[base + GG + g]       + b_hh[GG + g];
        float in_ = (float)gI[base + 2 * GG + g]  + b_ih[2 * GG + g];
        float hn = (float)gH[base + 2 * GG + g]   + b_hh[2 * GG + g];
        float r = 1.f / (1.f + expf(-(ir + hr)));
        float z = 1.f / (1.f + expf(-(iz + hz)));
        float nn = tanhf(in_ + r * hn);
        float hv = (float)h16[(size_t)m * GG + g];
        comb[j] = (1.f - z) * nn + z * hv;
    }
    float mean = block_sum(comb[0] + comb[1], red) * (1.f / 512.f);
    float d0 = comb[0] - mean, d1 = comb[1] - mean;
    float var = block_sum(d0 * d0 + d1 * d1, red) * (1.f / 512.f);
    float rstd = rsqrtf(var + 1e-5f);
    x16[(size_t)m * GG + t]       = (_Float16)(d0 * rstd * ln_g[t] + ln_b[t]);
    x16[(size_t)m * GG + t + 256] = (_Float16)(d1 * rstd * ln_g[t + 256] + ln_b[t + 256]);
}

// ---------------- fused pre-LN + score head ----------------
__global__ __launch_bounds__(256) void ln_score(
    const float* __restrict__ y3, const float* __restrict__ ln_g, const float* __restrict__ ln_b,
    const float* __restrict__ Ws, const float* __restrict__ bs, float* __restrict__ scores)
{
    const int m = blockIdx.x, t = threadIdx.x;
    __shared__ float red[256];
    float v0 = y3[(size_t)m * DD + t];
    float v1 = y3[(size_t)m * DD + t + 256];
    float mean = block_sum(v0 + v1, red) * (1.f / 512.f);
    float d0 = v0 - mean, d1 = v1 - mean;
    float var = block_sum(d0 * d0 + d1 * d1, red) * (1.f / 512.f);
    float rstd = rsqrtf(var + 1e-5f);
    float p = (d0 * rstd * ln_g[t] + ln_b[t]) * Ws[t]
            + (d1 * rstd * ln_g[t + 256] + ln_b[t + 256]) * Ws[t + 256];
    float s = block_sum(p, red);
    if (t == 0) scores[m] = s + bs[0];
}

// ---------------- per-batch softmax over S=1024 ----------------
__global__ __launch_bounds__(256) void softmax_b(
    const float* __restrict__ scores, float* __restrict__ probs)
{
    const int b = blockIdx.x, t = threadIdx.x;
    __shared__ float red[256];
    float x[4];
    #pragma unroll
    for (int j = 0; j < 4; ++j) x[j] = scores[b * SS + t + j * 256];
    float mx = fmaxf(fmaxf(x[0], x[1]), fmaxf(x[2], x[3]));
    mx = block_max(mx, red);
    float e[4], ls = 0.f;
    #pragma unroll
    for (int j = 0; j < 4; ++j) { e[j] = expf(x[j] - mx); ls += e[j]; }
    float sum = block_sum(ls, red);
    float inv = 1.f / sum;
    #pragma unroll
    for (int j = 0; j < 4; ++j) probs[b * SS + t + j * 256] = e[j] * inv;
}

// ---------------- sparse scatter into spectrum ----------------
__global__ void scatter_spect(const int* __restrict__ midx, const float* __restrict__ inten,
                              const float* __restrict__ probs, float* __restrict__ spect)
{
    const int i = blockIdx.x * 256 + threadIdx.x;
    if (i >= BB * SS * MM) return;
    const int b = i >> 13;            // S*M = 8192
    const int s = (i >> 3) & (SS - 1);
    atomicAdd(spect + b * NBINS + midx[i], inten[i] * probs[b * SS + s]);
}

// ---------------- host launch ----------------
extern "C" void kernel_launch(void* const* d_in, const int* in_sizes, int n_in,
                              void* d_out, int out_size, void* d_ws, size_t ws_size,
                              hipStream_t stream) {
    const float* vert_feat = (const float*)d_in[0];
    const float* vert_mask = (const float*)d_in[1];
    const float* elem_oh   = (const float*)d_in[2];
    const int*   subs      = (const int*)d_in[3];
    const int*   peaks_idx = (const int*)d_in[4];
    const float* peaks_int = (const float*)d_in[5];
    const float* ln_sub_g  = (const float*)d_in[6];
    const float* ln_sub_b  = (const float*)d_in[7];
    const float* W_ih      = (const float*)d_in[8];
    const float* W_hh      = (const float*)d_in[9];
    const float* b_ih      = (const float*)d_in[10];
    const float* b_hh      = (const float*)d_in[11];
    const float* ln_post_g = (const float*)d_in[12];
    const float* ln_post_b = (const float*)d_in[13];
    const float* W1        = (const float*)d_in[14];
    const float* b1        = (const float*)d_in[15];
    const float* W2a       = (const float*)d_in[16];
    const float* b2a       = (const float*)d_in[17];
    const float* W2b       = (const float*)d_in[18];
    const float* b2b       = (const float*)d_in[19];
    const float* ln_pre_g  = (const float*)d_in[20];
    const float* ln_pre_b  = (const float*)d_in[21];
    const float* W_s       = (const float*)d_in[22];
    const float* b_s       = (const float*)d_in[23];

    // workspace bump allocation
    char* ws = (char*)d_ws;
    size_t off = 0;
    auto alloc = [&](size_t bytes) -> void* {
        void* p = ws + off;
        off += (bytes + 255) & ~(size_t)255;
        return p;
    };
    _Float16* wih16  = (_Float16*)alloc((size_t)G3 * FEPAD * 2);
    _Float16* whh16  = (_Float16*)alloc((size_t)G3 * GG * 2);
    _Float16* w1_16  = (_Float16*)alloc((size_t)DD * GG * 2);
    _Float16* w2a16  = (_Float16*)alloc((size_t)DD * DD * 2);
    _Float16* w2b16  = (_Float16*)alloc((size_t)DD * DD * 2);
    _Float16* h16    = (_Float16*)alloc((size_t)MROWS * GG * 2);
    _Float16* form16 = (_Float16*)alloc((size_t)MROWS * FEPAD * 2);
    _Float16* gatesI = (_Float16*)alloc((size_t)MROWS * G3 * 2);
    _Float16* gatesH = (_Float16*)alloc((size_t)MROWS * G3 * 2);
    _Float16* x16    = (_Float16*)alloc((size_t)MROWS * GG * 2);
    _Float16* y1_16  = (_Float16*)alloc((size_t)MROWS * DD * 2);
    _Float16* y2_16  = (_Float16*)alloc((size_t)MROWS * DD * 2);
    float*    y3_32  = (float*)   alloc((size_t)MROWS * DD * 4);
    float*    scores = (float*)   alloc((size_t)MROWS * 4);

    float* out   = (float*)d_out;
    float* spect = out;                  // B*NBINS
    float* probs = out + BB * NBINS;     // B*S

    // 1) convert / pad weights to f16
    cvt_pad_f16<<<(G3 * FEPAD + 255) / 256, 256, 0, stream>>>(W_ih, wih16, G3, EE * OHH, FEPAD);
    cvt_pad_f16<<<(G3 * GG + 255) / 256, 256, 0, stream>>>(W_hh, whh16, G3, GG, GG);
    cvt_pad_f16<<<(DD * GG + 255) / 256, 256, 0, stream>>>(W1, w1_16, DD, GG, GG);
    cvt_pad_f16<<<(DD * DD + 255) / 256, 256, 0, stream>>>(W2a, w2a16, DD, DD, DD);
    cvt_pad_f16<<<(DD * DD + 255) / 256, 256, 0, stream>>>(W2b, w2b16, DD, DD, DD);
    zero_f32<<<(BB * NBINS + 255) / 256, 256, 0, stream>>>(spect, BB * NBINS);

    // 2) subset pooling + LN + formula encoding
    pool_ln_formula<<<MROWS, 256, 0, stream>>>(vert_feat, vert_mask, elem_oh, subs,
                                               ln_sub_g, ln_sub_b, h16, form16);

    // 3) GRU gate GEMMs (WMMA): gi = formulae * W_ih^T ; gh = h * W_hh^T
    gemm16x4<<<dim3(MROWS / 16, G3 / 256), 128, 0, stream>>>(form16, wih16, nullptr, gatesI,
                                                             nullptr, G3, FEPAD, 0);
    gemm16x4<<<dim3(MROWS / 16, G3 / 256), 128, 0, stream>>>(h16, whh16, nullptr, gatesH,
                                                             nullptr, G3, GG, 0);

    // 4) GRU elementwise + post LN
    gru_ln<<<MROWS, 256, 0, stream>>>(gatesI, gatesH, h16, b_ih, b_hh,
                                      ln_post_g, ln_post_b, x16);

    // 5) MLP (WMMA + fused bias/ReLU)
    gemm16x4<<<dim3(MROWS / 16, DD / 256), 128, 0, stream>>>(x16, w1_16, nullptr, y1_16, b1, DD, GG, 1);
    gemm16x4<<<dim3(MROWS / 16, DD / 256), 128, 0, stream>>>(y1_16, w2a16, nullptr, y2_16, b2a, DD, DD, 1);
    gemm16x4<<<dim3(MROWS / 16, DD / 256), 128, 0, stream>>>(y2_16, w2b16, y3_32, nullptr, b2b, DD, DD, 1);

    // 6) pre-LN + score head, softmax, scatter
    ln_score<<<MROWS, 256, 0, stream>>>(y3_32, ln_pre_g, ln_pre_b, W_s, b_s, scores);
    softmax_b<<<BB, 256, 0, stream>>>(scores, probs);
    scatter_spect<<<(BB * SS * MM + 255) / 256, 256, 0, stream>>>(peaks_idx, peaks_int, probs, spect);

    (void)in_sizes; (void)n_in; (void)out_size; (void)ws_size;
}
